// MPNN_85341000171713
// MI455X (gfx1250) — matrix operand order repaired and checked by reading, not compiled, not run on hardware
//
#include <hip/hip_runtime.h>
#include <hip/hip_bf16.h>

// ---------------------------------------------------------------------------
// MPNN (SchNet-style) for MI455X / gfx1250 (wave32, WMMA bf16 -> f32)
//
// All GEMMs computed TRANSPOSED (weights = A matrix, activations = B matrix)
// so the WMMA C-fragment gives each lane ONE row (node/edge) x 8 consecutive
// features:
//   - weights pre-swizzled into per-lane fragment order -> 32B contiguous
//     global loads per lane (1KB/wave, fully coalesced)
//   - activation B-fragments = one contiguous 32B LDS chunk per lane
//   - epilogue: packed v8bf (16B) stores / gathers, contiguous f32 atomics
// ---------------------------------------------------------------------------

#define NN_  50000
#define EE_  1600000
#define GG_  500
#define HH_  128
#define NGP_ 64      // NG=50 padded to 64 (zero pad)
#define LL_  6

typedef __attribute__((ext_vector_type(16))) __bf16 v16bf;
typedef __attribute__((ext_vector_type(8)))  __bf16 v8bf;
typedef __attribute__((ext_vector_type(8)))  float  v8f;

__device__ __forceinline__ float sspf(float x) {
    // softplus(x) - log(2), numerically stable
    return fmaxf(x, 0.f) + log1pf(__expf(-fabsf(x))) - 0.6931471805599453f;
}

// 16-bit A-matrix 16x32 lane/half -> K mapping (contiguous, matches B): used when
// pre-swizzling weight matrices into fragment order.
__device__ __forceinline__ int kmapA(int hh, int lane) {
    int k = hh;                             // halves 0-15 -> K0-15
    return k + ((lane & 16) ? 16 : 0);      // upper half-wave shifted by 16
}
// 16-bit B-matrix 32x16, order chosen here: lane halves 0..15 -> K = hh + (upper-half ? 16 : 0)
// (contiguous 32B chunk per lane -- exploited for LDS b128 reads)

__device__ __forceinline__ v8f wmma_bf16(v16bf a, v16bf b, v8f c) {
    return __builtin_amdgcn_wmma_f32_16x16x32_bf16(false, a, false, b,
                                                   (short)0, c, false, false);
}

__device__ __forceinline__ void atomic_add_f32(float* p, float v) {
    __hip_atomic_fetch_add(p, v, __ATOMIC_RELAXED, __HIP_MEMORY_SCOPE_AGENT);
}

// ------------------------- small utility kernels ---------------------------

__global__ void zerof_kernel(float* p, int n) {
    int i = blockIdx.x * 256 + threadIdx.x;
    if (i < n) p[i] = 0.f;
}

// v[n][0:128] = bf16(emb[z[n]][0:128]), vectorized 8 elements/thread
__global__ void embed_kernel(const int* z, const float* emb, __bf16* v) {
    int i = blockIdx.x * 256 + threadIdx.x;           // NN_*16 chunks
    int n = i >> 4, off = (i & 15) * 8;
    const float4* p = (const float4*)&emb[z[n] * HH_ + off];
    float4 x0 = p[0], x1 = p[1];
    v8bf o;
    o[0] = (__bf16)x0.x; o[1] = (__bf16)x0.y; o[2] = (__bf16)x0.z; o[3] = (__bf16)x0.w;
    o[4] = (__bf16)x1.x; o[5] = (__bf16)x1.y; o[6] = (__bf16)x1.z; o[7] = (__bf16)x1.w;
    *(v8bf*)&v[(long)n * HH_ + off] = o;
}

// Pre-swizzle weights into per-lane A-fragment order (transposed GEMMs):
//   WnS/WoS: [l][ftile(8)][ks(4)][lane(32)][hh(16)]  <- W[l][k][f]
//   WeS:     [l][ftile(8)][ks(2)][lane(32)][hh(16)]  <- We[l][k][f], k>=50 -> 0
// with k = ks*32 + kmapA(hh,lane), f = ftile*16 + (lane&15).
__global__ void convert_params_kernel(const float* Wn, const float* We, const float* Wo,
                                      __bf16* WnS, __bf16* WoS, __bf16* WeS) {
    int i = blockIdx.x * 256 + threadIdx.x;
    if (i < LL_ * HH_ * HH_) {                         // 98304
        int hh = i & 15, lane = (i >> 4) & 31, ks = (i >> 9) & 3;
        int ft = (i >> 11) & 7, l = i >> 14;
        int k = ks * 32 + kmapA(hh, lane);
        int f = ft * 16 + (lane & 15);
        WnS[i] = (__bf16)Wn[((l * HH_) + k) * HH_ + f];
        WoS[i] = (__bf16)Wo[((l * HH_) + k) * HH_ + f];
    }
    if (i < LL_ * NGP_ * HH_) {                        // 49152
        int hh = i & 15, lane = (i >> 4) & 31, ks = (i >> 9) & 1;
        int ft = (i >> 10) & 7, l = i >> 13;
        int k = ks * 32 + kmapA(hh, lane);
        int f = ft * 16 + (lane & 15);
        WeS[i] = (k < 50) ? (__bf16)We[((l * 50) + k) * HH_ + f] : (__bf16)0.f;
    }
}

// ------------------- transposed dense WMMA GEMM (K=128) --------------------
// Out[n][f] = act( sum_k Act[n][k] * W[k][f] + bias[f] ), computed as
// Wt(16f x 32k) x Act^T(32k x 16n) tiles. Block = 256 thr = 8 waves = 8 ftiles
// over one 16-node strip; 4 WMMA K-steps. Lane result: node (lane&15),
// features f0..f0+7 -> one packed 16B store.

template <bool AF32, bool ACT>
__global__ void gemmT128_kernel(const void* Act_in, const __bf16* __restrict__ Wsw,
                                const float* __restrict__ bias, __bf16* __restrict__ Out) {
    __shared__ __bf16 sV[16][144];      // 16 nodes x 128 k, 288B stride (32B mult)
    __shared__ float  sBias[HH_];
    const int tid  = threadIdx.x;
    const int lane = tid & 31;
    const int wave = tid >> 5;
    const long n0  = (long)blockIdx.x * 16;

    {   // stage activations: one 8-half chunk per thread (256 chunks total)
        int node = tid >> 4, off = (tid & 15) * 8;
        if constexpr (AF32) {
            const float4* p = (const float4*)((const float*)Act_in + (n0 + node) * HH_ + off);
            float4 x0 = p[0], x1 = p[1];
            v8bf o;
            o[0] = (__bf16)x0.x; o[1] = (__bf16)x0.y; o[2] = (__bf16)x0.z; o[3] = (__bf16)x0.w;
            o[4] = (__bf16)x1.x; o[5] = (__bf16)x1.y; o[6] = (__bf16)x1.z; o[7] = (__bf16)x1.w;
            *(v8bf*)&sV[node][off] = o;
        } else {
            *(v8bf*)&sV[node][off] =
                *(const v8bf*)((const __bf16*)Act_in + (n0 + node) * HH_ + off);
        }
    }
    if (tid < HH_) sBias[tid] = bias ? bias[tid] : 0.f;
    __syncthreads();

    const int node = lane & 15;
    const int koff = (lane & 16) ? 16 : 0;
    const v16bf* Wp = (const v16bf*)Wsw + (size_t)wave * 4 * 32;  // ftile slice

    v8f acc = {};
#pragma unroll
    for (int ks = 0; ks < 4; ++ks) {
        v16bf a = Wp[ks * 32 + lane];                              // 32B coalesced
        v16bf b = *(const v16bf*)&sV[node][ks * 32 + koff];        // 32B LDS chunk
        acc = wmma_bf16(a, b, acc);
    }

    const int f0 = wave * 16 + ((lane & 16) ? 8 : 0);
    v8bf outv;
#pragma unroll
    for (int j = 0; j < 8; ++j) {
        float x = acc[j] + sBias[f0 + j];
        if (ACT) x = sspf(x);
        outv[j] = (__bf16)x;
    }
    *(v8bf*)&Out[(n0 + node) * HH_ + f0] = outv;
}

// ------------------------- edge gate + gather + scatter --------------------
// gate^T = WeP^T(16f x 32k) x e^T(32k x 16edges); lane result = one edge x
// 8 consecutive features -> one v8bf gather of h[col], 8 contiguous f32
// atomics into agg[row]. Block = 8 waves x 16 edges = 128 edges.

__global__ void edge_msg_kernel(const float* __restrict__ pos, const int* __restrict__ ei,
                                const float* __restrict__ distW, const float* __restrict__ distb,
                                const __bf16* __restrict__ WeS, const float* __restrict__ be,
                                const __bf16* __restrict__ h, float* __restrict__ agg, int E) {
    __shared__ float sW[NGP_], sB[NGP_], sBe[HH_];
    __shared__ int   sRow[128], sCol[128];
    __shared__ float sDist[128];

    const int tid  = threadIdx.x;
    const int lane = tid & 31;
    const int wave = tid >> 5;

    if (tid < NGP_) {
        sW[tid] = (tid < 50) ? distW[tid] : 0.f;
        sB[tid] = (tid < 50) ? distb[tid] : 0.f;
    }
    if (tid >= 128 && tid < 128 + HH_) sBe[tid - 128] = be[tid - 128];
    if (tid < 128) {
        long e = (long)blockIdx.x * 128 + tid;
        int rn = ei[e];
        int cn = ei[(long)E + e];
        float dx = pos[rn * 3 + 0] - pos[cn * 3 + 0];
        float dy = pos[rn * 3 + 1] - pos[cn * 3 + 1];
        float dz = pos[rn * 3 + 2] - pos[cn * 3 + 2];
        sRow[tid]  = rn;
        sCol[tid]  = cn;
        sDist[tid] = sqrtf(dx * dx + dy * dy + dz * dz);
    }
    __syncthreads();

    const int eloc = (wave << 4) + (lane & 15);       // this lane's edge
    const float d  = sDist[eloc];
    const long hbase = (long)sCol[eloc] * HH_;
    const long abase = (long)sRow[eloc] * HH_;

    // B fragments (edge features, two K=32 steps), contiguous k per lane
    const int koff = (lane & 16) ? 16 : 0;
    v16bf b0, b1;
#pragma unroll
    for (int hh = 0; hh < 16; ++hh) {
        int k = hh + koff;
        b0[hh] = (__bf16)(d * sW[k] + sB[k]);
        b1[hh] = (__bf16)(d * sW[k + 32] + sB[k + 32]);
    }

    const v16bf* Wp = (const v16bf*)WeS;
#pragma unroll
    for (int ftile = 0; ftile < 8; ++ftile) {
        v8f acc = {};
        acc = wmma_bf16(Wp[(ftile * 2 + 0) * 32 + lane], b0, acc);
        acc = wmma_bf16(Wp[(ftile * 2 + 1) * 32 + lane], b1, acc);

        const int f0 = ftile * 16 + ((lane & 16) ? 8 : 0);
        v8bf hv = *(const v8bf*)&h[hbase + f0];        // single 16B gather
#pragma unroll
        for (int j = 0; j < 8; ++j) {
            float m = (acc[j] + sBe[f0 + j]) * (float)hv[j];
            atomic_add_f32(&agg[abase + f0 + j], m);   // contiguous addresses
        }
    }
}

// ------------------------- readout -----------------------------------------
// One wave per node: lane computes hidden units j=lane and j=lane+32,
// wave32 shuffle reduction, one atomic per node into out[batch[n]].

__global__ void readout_kernel(const __bf16* __restrict__ v, const int* __restrict__ batch,
                               const float* __restrict__ W1, const float* __restrict__ b1,
                               const float* __restrict__ W2, const float* __restrict__ b2,
                               float* __restrict__ out) {
    const int lane = threadIdx.x & 31;
    const int wave = threadIdx.x >> 5;
    const long n   = (long)blockIdx.x * 8 + wave;

    float s0 = b1[lane];
    float s1 = b1[lane + 32];
#pragma unroll
    for (int c8 = 0; c8 < 16; ++c8) {
        v8bf vv = *(const v8bf*)&v[n * HH_ + c8 * 8];
#pragma unroll
        for (int j = 0; j < 8; ++j) {
            int c = c8 * 8 + j;
            float vf = (float)vv[j];
            s0 += vf * W1[c * 64 + lane];
            s1 += vf * W1[c * 64 + lane + 32];
        }
    }
    float p = sspf(s0) * W2[lane] + sspf(s1) * W2[lane + 32];
#pragma unroll
    for (int off = 16; off > 0; off >>= 1) p += __shfl_down(p, off, 32);
    if (lane == 0) atomic_add_f32(&out[batch[n]], p + b2[0]);
}

// ------------------------- host launcher -----------------------------------

extern "C" void kernel_launch(void* const* d_in, const int* in_sizes, int n_in,
                              void* d_out, int out_size, void* d_ws, size_t ws_size,
                              hipStream_t stream) {
    const int*   z     = (const int*)  d_in[0];
    const float* pos   = (const float*)d_in[1];
    const int*   batch = (const int*)  d_in[2];
    const int*   ei    = (const int*)  d_in[3];
    const float* emb   = (const float*)d_in[4];
    const float* distW = (const float*)d_in[5];
    const float* distb = (const float*)d_in[6];
    const float* Wn    = (const float*)d_in[7];
    const float* We    = (const float*)d_in[8];
    const float* be    = (const float*)d_in[9];
    const float* Wo    = (const float*)d_in[10];
    const float* bo    = (const float*)d_in[11];
    const float* W1    = (const float*)d_in[12];
    const float* b1    = (const float*)d_in[13];
    const float* W2    = (const float*)d_in[14];
    const float* b2    = (const float*)d_in[15];
    float* out = (float*)d_out;

    auto alignup = [](size_t x) { return (x + 255) & ~(size_t)255; };
    char* ws = (char*)d_ws;
    __bf16* v    = (__bf16*)ws; ws += alignup((size_t)NN_ * HH_ * 2);
    __bf16* h    = (__bf16*)ws; ws += alignup((size_t)NN_ * HH_ * 2);
    float*  agg  = (float*) ws; ws += alignup((size_t)NN_ * HH_ * 4);
    __bf16* WnS  = (__bf16*)ws; ws += alignup((size_t)LL_ * HH_ * HH_ * 2);
    __bf16* WoS  = (__bf16*)ws; ws += alignup((size_t)LL_ * HH_ * HH_ * 2);
    __bf16* WeS  = (__bf16*)ws; ws += alignup((size_t)LL_ * NGP_ * HH_ * 2);

    convert_params_kernel<<<(LL_ * HH_ * HH_ + 255) / 256, 256, 0, stream>>>(
        Wn, We, Wo, WnS, WoS, WeS);
    embed_kernel<<<NN_ * 16 / 256, 256, 0, stream>>>(z, emb, v);

    for (int l = 0; l < LL_; ++l) {
        gemmT128_kernel<false, false><<<NN_ / 16, 256, 0, stream>>>(
            v, WnS + (size_t)l * HH_ * HH_, (const float*)nullptr, h);
        zerof_kernel<<<(NN_ * HH_ + 255) / 256, 256, 0, stream>>>(agg, NN_ * HH_);
        edge_msg_kernel<<<EE_ / 128, 256, 0, stream>>>(
            pos, ei, distW, distb, WeS + (size_t)l * NGP_ * HH_, be + (size_t)l * HH_,
            h, agg, EE_);
        gemmT128_kernel<true, true><<<NN_ / 16, 256, 0, stream>>>(
            agg, WoS + (size_t)l * HH_ * HH_, bo + (size_t)l * HH_, v);
    }

    zerof_kernel<<<(GG_ + 255) / 256, 256, 0, stream>>>(out, GG_);
    readout_kernel<<<NN_ / 8, 256, 0, stream>>>(v, batch, W1, b1, W2, b2, out);
}
